// GNN_52020643889506
// MI455X (gfx1250) — matrix operand order repaired and checked by reading, not compiled, not run on hardware
//
#include <hip/hip_runtime.h>

#define NN   50000   // nodes
#define EE   200000  // edges
#define CC   32      // conv dim
#define ND   75      // atom features
#define NDP  96      // padded K for projection
#define EDF  12      // bond features
#define LL   3       // layers
#define GG   500     // graphs
#define HID  5       // message hidden
#define EPSV 1e-5f
#define TCOLS 224    // 5*32 (W2_k) + 32 (b2) + 32 (rootW)

typedef __attribute__((ext_vector_type(16))) _Float16 v16h;
typedef __attribute__((ext_vector_type(8)))  _Float16 v8h;
typedef __attribute__((ext_vector_type(8)))  float    v8f;

union AV16 { v16h v; _Float16 e[16]; v8h h2[2]; };

__device__ __forceinline__ float leakyf(float v) { return v > 0.f ? v : 0.01f * v; }

__device__ __forceinline__ void atomAddF(float* p, float v) {
    __hip_atomic_fetch_add(p, v, __ATOMIC_RELAXED, __HIP_MEMORY_SCOPE_AGENT);
}

// ---------------------------------------------------------------- pad x -> f16 [N, 96]
__global__ void pad_x_kernel(const float* __restrict__ x, _Float16* __restrict__ xh) {
    int gid = blockIdx.x * blockDim.x + threadIdx.x;
    if (gid >= NN * NDP) return;
    int n = gid / NDP, k = gid - n * NDP;
    xh[gid] = (_Float16)(k < ND ? x[n * ND + k] : 0.f);
}

// ---------------------------------------------------------------- pack lin_W into WMMA-B layout
// Bproj[(c*2+t)*32 + lane][j]  with  K = c*32 + (lane>>4)*16 + j, col = t*16 + (lane&15)
__global__ void pack_proj_kernel(const float* __restrict__ linW, _Float16* __restrict__ Bproj) {
    int gid = blockIdx.x * blockDim.x + threadIdx.x;
    if (gid >= 6 * 32) return;
    int tile = gid >> 5, lane = gid & 31;
    int c = tile >> 1, t = tile & 1;
    int col = t * 16 + (lane & 15);
    int hi = lane >> 4;
    _Float16* dst = Bproj + (size_t)gid * 16;
#pragma unroll
    for (int j = 0; j < 16; ++j) {
        int K = c * 32 + hi * 16 + j;
        dst[j] = (_Float16)(K < ND ? linW[K * CC + col] : 0.f);
    }
}

// ---------------------------------------------------------------- pack [W2_0..4 | b2 | rootW] (all layers) into WMMA-B layout
// Bcat[((l*14)+t)*32 + lane][j]  with  K(c) = (lane>>4)*16 + j, col = t*16 + (lane&15)
__global__ void pack_cat_kernel(const float* __restrict__ mesW2,  // [L,5,1024]
                                const float* __restrict__ mesb2,  // [L,1024]
                                const float* __restrict__ rootW,  // [L,32,32]
                                _Float16* __restrict__ Bcat) {
    int gid = blockIdx.x * blockDim.x + threadIdx.x;
    if (gid >= LL * 14 * 32) return;
    int tl = gid >> 5, lane = gid & 31;
    int l = tl / 14, t = tl - l * 14;
    int col = t * 16 + (lane & 15);
    int hi = lane >> 4;
    const float* W2 = mesW2 + (size_t)l * HID * CC * CC;
    const float* b2 = mesb2 + (size_t)l * CC * CC;
    const float* rt = rootW + (size_t)l * CC * CC;
    _Float16* dst = Bcat + (size_t)gid * 16;
#pragma unroll
    for (int j = 0; j < 16; ++j) {
        int c = hi * 16 + j;
        float w;
        if (t < 10)      { int k = col >> 5, d = col & 31; w = W2[k * 1024 + c * 32 + d]; }
        else if (t < 12) { int d = col - 160;              w = b2[c * 32 + d]; }
        else             { int d = col - 192;              w = rt[c * 32 + d]; }
        dst[j] = (_Float16)w;
    }
}

// ---------------------------------------------------------------- h = leaky(x @ lin_W + lin_b); also f16 mirror
__global__ void proj_kernel(const _Float16* __restrict__ xh,
                            const _Float16* __restrict__ Bproj,
                            const float* __restrict__ linb,
                            float* __restrict__ h, _Float16* __restrict__ h16) {
    int wave = (blockIdx.x * blockDim.x + threadIdx.x) >> 5;
    int lane = threadIdx.x & 31;
    if (wave >= NN / 16) return;
    int m = lane & 15, hi = lane >> 4;
    int node = wave * 16 + m;

    v8f acc0 = {}; v8f acc1 = {};
#pragma unroll
    for (int c = 0; c < 3; ++c) {
        AV16 a;
        const _Float16* ap = xh + (size_t)node * NDP + c * 32 + hi * 16;
        a.h2[0] = *(const v8h*)(ap);
        a.h2[1] = *(const v8h*)(ap + 8);
#pragma unroll
        for (int t = 0; t < 2; ++t) {
            AV16 b;
            const _Float16* bp = Bproj + (size_t)((c * 2 + t) * 32 + lane) * 16;
            b.h2[0] = *(const v8h*)(bp);
            b.h2[1] = *(const v8h*)(bp + 8);
            if (t == 0)
                acc0 = __builtin_amdgcn_wmma_f32_16x16x32_f16(false, a.v, false, b.v,
                                                              (short)0, acc0, false, false);
            else
                acc1 = __builtin_amdgcn_wmma_f32_16x16x32_f16(false, a.v, false, b.v,
                                                              (short)0, acc1, false, false);
        }
    }
#pragma unroll
    for (int r = 0; r < 8; ++r) {
        int row = wave * 16 + hi * 8 + r;
        float v0 = leakyf(acc0[r] + linb[m]);
        float v1 = leakyf(acc1[r] + linb[16 + m]);
        h[(size_t)row * CC + m]        = v0;
        h[(size_t)row * CC + 16 + m]   = v1;
        h16[(size_t)row * CC + m]      = (_Float16)v0;
        h16[(size_t)row * CC + 16 + m] = (_Float16)v1;
    }
}

// ---------------------------------------------------------------- T = h @ [W2_0..W2_4 | b2 | rootW]  -> [N, 224]
__global__ void tmat_kernel(const _Float16* __restrict__ h16,
                            const _Float16* __restrict__ Bcat_l,  // 14 packed tiles
                            float* __restrict__ T) {
    int wave = (blockIdx.x * blockDim.x + threadIdx.x) >> 5;
    int lane = threadIdx.x & 31;
    if (wave >= NN / 16) return;
    int m = lane & 15, hi = lane >> 4;
    int node = wave * 16 + m;

    AV16 a;
    const _Float16* ap = h16 + (size_t)node * CC + hi * 16;
    a.h2[0] = *(const v8h*)(ap);
    a.h2[1] = *(const v8h*)(ap + 8);

#pragma unroll
    for (int t = 0; t < 14; ++t) {
        AV16 b;
        const _Float16* bp = Bcat_l + (size_t)(t * 32 + lane) * 16;
        b.h2[0] = *(const v8h*)(bp);
        b.h2[1] = *(const v8h*)(bp + 8);
        v8f acc = {};
        acc = __builtin_amdgcn_wmma_f32_16x16x32_f16(false, a.v, false, b.v,
                                                     (short)0, acc, false, false);
        int col = t * 16 + m;
#pragma unroll
        for (int r = 0; r < 8; ++r) {
            int row = wave * 16 + hi * 8 + r;
            T[(size_t)row * TCOLS + col] = acc[r];
        }
    }
}

// ---------------------------------------------------------------- per-edge message + scatter-add
__global__ void edge_kernel(const float* __restrict__ T,
                            const float* __restrict__ edge_attr,
                            const int*   __restrict__ edge_index,
                            const float* __restrict__ W1,  // [12,5] this layer
                            const float* __restrict__ b1,  // [5]
                            float* __restrict__ agg) {
    int e    = (blockIdx.x * blockDim.x + threadIdx.x) >> 5;
    int lane = threadIdx.x & 31;
    if (e >= EE) return;

    float rw[HID];
#pragma unroll
    for (int k = 0; k < HID; ++k) {
        float s = b1[k];
#pragma unroll
        for (int j = 0; j < EDF; ++j) s = fmaf(edge_attr[(size_t)e * EDF + j], W1[j * HID + k], s);
        rw[k] = s > 0.f ? s : 0.f;
    }
    int src = edge_index[e];
    int dst = edge_index[EE + e];
    const float* trow = T + (size_t)src * TCOLS;
    float msg = trow[160 + lane];  // b2 contribution
#pragma unroll
    for (int k = 0; k < HID; ++k) msg = fmaf(rw[k], trow[k * 32 + lane], msg);
    atomAddF(&agg[(size_t)dst * CC + lane], msg);
}

// ---------------------------------------------------------------- out_pre = agg + h@rootW + conv_b ; BN partial sums
#define NPW 64
__global__ void post_kernel(const float* __restrict__ agg,
                            const float* __restrict__ T,
                            const float* __restrict__ convb,
                            float* __restrict__ outpre,
                            float* __restrict__ bnsum, float* __restrict__ bnsq) {
    int wave = (blockIdx.x * blockDim.x + threadIdx.x) >> 5;
    int lane = threadIdx.x & 31;
    int base = wave * NPW;
    if (base >= NN) return;
    float s = 0.f, sq = 0.f;
    float cb = convb[lane];
    for (int i = 0; i < NPW; ++i) {
        int n = base + i;
        if (n >= NN) break;
        float v = agg[(size_t)n * CC + lane] + T[(size_t)n * TCOLS + 192 + lane] + cb;
        outpre[(size_t)n * CC + lane] = v;
        s += v; sq += v * v;
    }
    atomAddF(&bnsum[lane], s);
    atomAddF(&bnsq[lane], sq);
}

// ---------------------------------------------------------------- normalize + (leaky) -> h (f32 + f16 mirror)
__global__ void bn_kernel(const float* __restrict__ outpre,
                          const float* __restrict__ bnsum, const float* __restrict__ bnsq,
                          const float* __restrict__ g, const float* __restrict__ b,
                          float* __restrict__ h, _Float16* __restrict__ h16, int do_leaky) {
    int gid = blockIdx.x * blockDim.x + threadIdx.x;
    if (gid >= NN * CC) return;
    int c = gid & 31;
    float mu  = bnsum[c] * (1.f / NN);
    float var = bnsq[c] * (1.f / NN) - mu * mu;
    float v = (outpre[gid] - mu) * rsqrtf(var + EPSV) * g[c] + b[c];
    if (do_leaky) v = leakyf(v);
    h[gid]   = v;
    h16[gid] = (_Float16)v;
}

// ---------------------------------------------------------------- readout
__global__ void readout_init_kernel(const float* __restrict__ predb, float* __restrict__ out) {
    int gid = blockIdx.x * blockDim.x + threadIdx.x;
    if (gid < GG) out[gid] = predb[0];
}
__global__ void readout_kernel(const float* __restrict__ h, const int* __restrict__ batch,
                               const float* __restrict__ predW, float* __restrict__ out) {
    int n = blockIdx.x * blockDim.x + threadIdx.x;
    if (n >= NN) return;
    float s = 0.f;
#pragma unroll
    for (int c = 0; c < CC; ++c) s = fmaf(h[(size_t)n * CC + c], predW[c], s);
    atomAddF(&out[batch[n]], s);
}

// ================================================================ launch
extern "C" void kernel_launch(void* const* d_in, const int* in_sizes, int n_in,
                              void* d_out, int out_size, void* d_ws, size_t ws_size,
                              hipStream_t stream) {
    const float* x         = (const float*)d_in[0];
    const int*   edgeidx   = (const int*)  d_in[1];
    const float* edge_attr = (const float*)d_in[2];
    const int*   batch     = (const int*)  d_in[3];
    const float* linW      = (const float*)d_in[4];
    const float* linb      = (const float*)d_in[5];
    const float* mesW1     = (const float*)d_in[6];
    const float* mesb1     = (const float*)d_in[7];
    const float* mesW2     = (const float*)d_in[8];
    const float* mesb2     = (const float*)d_in[9];
    const float* rootW     = (const float*)d_in[10];
    const float* convb     = (const float*)d_in[11];
    const float* bng       = (const float*)d_in[12];
    const float* bnb       = (const float*)d_in[13];
    const float* predW     = (const float*)d_in[14];
    const float* predb     = (const float*)d_in[15];
    float* out = (float*)d_out;

    size_t off = 0;
    auto carve = [&](size_t bytes) {
        void* p = (char*)d_ws + off;
        off += (bytes + 255) & ~(size_t)255;
        return p;
    };
    _Float16* xh    = (_Float16*)carve((size_t)NN * NDP * 2);
    _Float16* h16   = (_Float16*)carve((size_t)NN * CC * 2);
    _Float16* Bproj = (_Float16*)carve((size_t)6 * 32 * 16 * 2);
    _Float16* Bcat  = (_Float16*)carve((size_t)LL * 14 * 32 * 16 * 2);
    float* h        = (float*)carve((size_t)NN * CC * 4);
    float* T        = (float*)carve((size_t)NN * TCOLS * 4);
    float* agg      = (float*)carve((size_t)NN * CC * 4);
    float* outpre   = (float*)carve((size_t)NN * CC * 4);
    float* bnstats  = (float*)carve(64 * 4);   // [0:32]=sum, [32:64]=sumsq

    const int TB = 256;
    int waves16 = NN / 16;  // 3125

    pad_x_kernel<<<(NN * NDP + TB - 1) / TB, TB, 0, stream>>>(x, xh);
    pack_proj_kernel<<<1, 192, 0, stream>>>(linW, Bproj);
    pack_cat_kernel<<<(LL * 14 * 32 + TB - 1) / TB, TB, 0, stream>>>(mesW2, mesb2, rootW, Bcat);

    proj_kernel<<<(waves16 * 32 + TB - 1) / TB, TB, 0, stream>>>(xh, Bproj, linb, h, h16);

    for (int l = 0; l < LL; ++l) {
        hipMemsetAsync(agg, 0, (size_t)NN * CC * 4, stream);
        hipMemsetAsync(bnstats, 0, 64 * 4, stream);

        tmat_kernel<<<(waves16 * 32 + TB - 1) / TB, TB, 0, stream>>>(
            h16, Bcat + (size_t)l * 14 * 32 * 16, T);

        edge_kernel<<<((size_t)EE * 32 + TB - 1) / TB, TB, 0, stream>>>(
            T, edge_attr, edgeidx, mesW1 + (size_t)l * EDF * HID, mesb1 + (size_t)l * HID, agg);

        int pwaves = (NN + NPW - 1) / NPW;
        post_kernel<<<(pwaves * 32 + TB - 1) / TB, TB, 0, stream>>>(
            agg, T, convb + (size_t)l * CC, outpre, bnstats, bnstats + 32);

        bn_kernel<<<(NN * CC + TB - 1) / TB, TB, 0, stream>>>(
            outpre, bnstats, bnstats + 32, bng + (size_t)l * CC, bnb + (size_t)l * CC,
            h, h16, (l < LL - 1) ? 1 : 0);
    }

    readout_init_kernel<<<(GG + TB - 1) / TB, TB, 0, stream>>>(predb, out);
    readout_kernel<<<(NN + TB - 1) / TB, TB, 0, stream>>>(h, batch, predW, out);
}
